// SparseMoE_Cross_Attention_48052094107927
// MI455X (gfx1250) — matrix operand order repaired and checked by reading, not compiled, not run on hardware
//
#include <hip/hip_runtime.h>
#include <hip/hip_bf16.h>

// ---------------------------------------------------------------------------
// SparseMoE + per-token cross-head attention for MI455X (gfx1250, wave32).
// Heavy GEMMs use V_WMMA_F32_16X16X32_F16 (f16 A/B, f32 accum), M=32 tiles,
// global_prefetch_b8 on the W stream for the next K-tile.
// ---------------------------------------------------------------------------

typedef __attribute__((ext_vector_type(16))) _Float16 v16h;
typedef __attribute__((ext_vector_type(8)))  float    v8f;
typedef __attribute__((ext_vector_type(4)))  unsigned int u32x4;
typedef __attribute__((ext_vector_type(4)))  float    f32x4;

union Frag16 { u32x4 u[2]; v16h h; };

#define BSZ     4096
#define DIMK    2048
#define NEXP    8
#define NHEAD   16
#define HDIM    128
#define NCOL    6144            // 3*DIM
#define MAX_SEG 264             // ceil((2*4096 + 8*31)/32)
#define MAX_SLOTS (MAX_SEG*32)
#define ATT_SCALE 0.08838834764831845f  // 1/sqrt(128)

// ---------------------------------------------------------------- zero init
__global__ void zero_kernel(float* __restrict__ p, long long n) {
    long long i = (long long)blockIdx.x * blockDim.x + threadIdx.x;
    long long stride = (long long)gridDim.x * blockDim.x;
    for (; i < n; i += stride) p[i] = 0.0f;
}

// ------------------------------------------------------- gating + top-2
__global__ __launch_bounds__(256)
void gate_kernel(const float* __restrict__ x, const float* __restrict__ Wg,
                 const float* __restrict__ bg, int* __restrict__ idx,
                 float* __restrict__ wgt, int* __restrict__ counts) {
    const int b = blockIdx.x;
    const int t = threadIdx.x;
    float part[NEXP];
#pragma unroll
    for (int e = 0; e < NEXP; ++e) part[e] = 0.0f;
    const float* xr = x + (size_t)b * DIMK;
    for (int d = t; d < DIMK; d += 256) {
        const float xv = xr[d];
#pragma unroll
        for (int e = 0; e < NEXP; ++e) part[e] += xv * Wg[d * NEXP + e];
    }
    __shared__ float sp[256 * NEXP];
#pragma unroll
    for (int e = 0; e < NEXP; ++e) sp[t * NEXP + e] = part[e];
    __syncthreads();
    __shared__ float logit[NEXP];
    if (t < NEXP) {
        float s = 0.0f;
        for (int i = 0; i < 256; ++i) s += sp[i * NEXP + t];
        logit[t] = s + bg[t];
    }
    __syncthreads();
    if (t == 0) {
        float mx = logit[0];
#pragma unroll
        for (int e = 1; e < NEXP; ++e) mx = fmaxf(mx, logit[e]);
        float ex[NEXP]; float sum = 0.0f;
#pragma unroll
        for (int e = 0; e < NEXP; ++e) { ex[e] = __expf(logit[e] - mx); sum += ex[e]; }
        const float inv = 1.0f / sum;
        int i1 = 0;
#pragma unroll
        for (int e = 1; e < NEXP; ++e) if (ex[e] > ex[i1]) i1 = e;
        int i2 = (i1 == 0) ? 1 : 0;
#pragma unroll
        for (int e = 0; e < NEXP; ++e) if (e != i1 && ex[e] > ex[i2]) i2 = e;
        idx[b * 2 + 0] = i1;  wgt[b * 2 + 0] = ex[i1] * inv;
        idx[b * 2 + 1] = i2;  wgt[b * 2 + 1] = ex[i2] * inv;
        atomicAdd(&counts[i1], 1);
        atomicAdd(&counts[i2], 1);
    }
}

// ------------------------------------------ per-expert padded offsets + segs
__global__ void offsets_kernel(const int* __restrict__ counts,
                               int* __restrict__ cursor,
                               int* __restrict__ pad_off,
                               int* __restrict__ seg_expert) {
    if (threadIdx.x == 0 && blockIdx.x == 0) {
        int po = 0;
        int lo[NEXP], hi[NEXP];
        for (int e = 0; e < NEXP; ++e) {
            pad_off[e] = po;
            lo[e] = po;
            int pc = (counts[e] + 31) & ~31;     // pad to M=32 tile
            po += pc;
            hi[e] = po;
            cursor[e] = 0;
        }
        const int total = po;
        for (int s = 0; s < MAX_SEG; ++s) {
            const int slot = s * 32;
            int ee = 0;
            if (slot < total) {
                for (int e = 0; e < NEXP; ++e)
                    if (slot >= lo[e] && slot < hi[e]) ee = e;
            }
            seg_expert[s] = ee;
        }
    }
}

// ------------------------------------------------ scatter tokens to experts
__global__ void scatter_kernel(const int* __restrict__ idx,
                               const float* __restrict__ wgt,
                               const int* __restrict__ pad_off,
                               int* __restrict__ cursor,
                               int* __restrict__ tok, float* __restrict__ wt) {
    const int b = blockIdx.x * blockDim.x + threadIdx.x;
    if (b < BSZ) {
#pragma unroll
        for (int j = 0; j < 2; ++j) {
            const int e = idx[b * 2 + j];
            const float w = wgt[b * 2 + j];
            const int pos = pad_off[e] + atomicAdd(&cursor[e], 1);
            tok[pos] = b;
            wt[pos] = w;
        }
    }
}

// --------------------------------------------------------------------------
// Grouped expert GEMM with WMMA f16.
// Block tile: M=32 tokens (one segment) x N=256 cols, K=2048.
// grid = (24 ntiles, MAX_SEG). ntile<8 -> y-pass (q cols), else x-pass (k,v).
// 8 waves/block; each wave owns a 2x2 grid of 16x16 output tiles
// (4 WMMAs per K-step, A/B fragments each reused twice).
// --------------------------------------------------------------------------
__global__ __launch_bounds__(256)
void moe_gemm_kernel(const float* __restrict__ x, const float* __restrict__ y,
                     const float* __restrict__ W,
                     const int* __restrict__ tok, const float* __restrict__ wt,
                     const int* __restrict__ seg_expert,
                     float* __restrict__ qbuf, float* __restrict__ kvbuf) {
    const int ntile = blockIdx.x;        // 0..23 over W cols (6144/256)
    const int seg   = blockIdx.y;        // token segment of 32
    const int e     = seg_expert[seg];
    const int colbase = ntile * 256;     // column base in W space

    const float* src;   float* dst;   int dstW, dcol0;
    if (ntile < 8) { src = y; dst = qbuf;  dstW = 2048; dcol0 = colbase; }
    else           { src = x; dst = kvbuf; dstW = 4096; dcol0 = colbase - 2048; }

    __shared__ __align__(16) _Float16 Ah[32 * 40];
    __shared__ __align__(16) _Float16 Bh[256 * 40];
    __shared__ int   tok_s[32];
    __shared__ float wt_s[32];

    const int t = threadIdx.x;
    if (t < 32) { tok_s[t] = tok[seg * 32 + t]; wt_s[t] = wt[seg * 32 + t]; }
    __syncthreads();

    const float* Wbase = W + (size_t)e * DIMK * NCOL + colbase;

    const int lane = t & 31;
    const int wv   = t >> 5;                 // wave id 0..7
    const int n0   = wv * 16;                // first N-subtile
    const int n1   = 128 + wv * 16;          // second N-subtile
    const int arow = lane & 15;              // A frag row within 16-row subtile
    const int akb  = (lane >> 4) * 8;        // A frag K base (0 or 8)
    const int bc0  = n0 + (lane & 15);       // B frag cols
    const int bc1  = n1 + (lane & 15);
    const int bk   = (lane >> 4) * 8;        // B frag K base (0 or 8)

    v8f acc00 = {}; v8f acc01 = {};          // rows 0..15  x (n0, n1)
    v8f acc10 = {}; v8f acc11 = {};          // rows 16..31 x (n0, n1)

    for (int k0 = 0; k0 < DIMK; k0 += 32) {
        // prefetch next K-tile of W: 32 rows x 1 cacheline(128B) per thread
        if (k0 + 32 < DIMK) {
            const int pr = t >> 3;              // 0..31  (K row)
            const int pc = (t & 7) * 32;        // 0..224 (col, 128B chunks)
            __builtin_prefetch(Wbase + (size_t)(k0 + 32 + pr) * NCOL + pc, 0, 1);
        }
        // stage A: 32 rows x 32 K (f32 -> f16), rows gathered via token ids
#pragma unroll
        for (int j = 0; j < 4; ++j) {
            const int i = t + j * 256;          // 0..1023
            const int r = i >> 5, kk = i & 31;
            Ah[r * 40 + kk] = (_Float16)src[(size_t)tok_s[r] * DIMK + k0 + kk];
        }
        // stage B: 32 K rows x 256 cols of W (f32 -> f16), stored col-major
#pragma unroll
        for (int i = 0; i < 8; ++i) {
            const int q  = i * 256 + t;          // float4 index, 0..2047
            const int kk = q >> 6;               // 64 float4 per K row
            const int c4 = (q & 63) << 2;
            const f32x4 v = *(const f32x4*)(Wbase + (size_t)(k0 + kk) * NCOL + c4);
            Bh[(c4 + 0) * 40 + kk] = (_Float16)v.x;
            Bh[(c4 + 1) * 40 + kk] = (_Float16)v.y;
            Bh[(c4 + 2) * 40 + kk] = (_Float16)v.z;
            Bh[(c4 + 3) * 40 + kk] = (_Float16)v.w;
        }
        __syncthreads();

        Frag16 a0, a1, b0, b1;
        a0.u[0] = *(const u32x4*)(&Ah[arow * 40 + akb]);
        a0.u[1] = *(const u32x4*)(&Ah[arow * 40 + akb + 16]);
        a1.u[0] = *(const u32x4*)(&Ah[(16 + arow) * 40 + akb]);
        a1.u[1] = *(const u32x4*)(&Ah[(16 + arow) * 40 + akb + 16]);
        b0.u[0] = *(const u32x4*)(&Bh[bc0 * 40 + bk]);
        b0.u[1] = *(const u32x4*)(&Bh[bc0 * 40 + bk + 16]);
        b1.u[0] = *(const u32x4*)(&Bh[bc1 * 40 + bk]);
        b1.u[1] = *(const u32x4*)(&Bh[bc1 * 40 + bk + 16]);

        acc00 = __builtin_amdgcn_wmma_f32_16x16x32_f16(
                    false, a0.h, false, b0.h, (short)0, acc00, false, false);
        acc01 = __builtin_amdgcn_wmma_f32_16x16x32_f16(
                    false, a0.h, false, b1.h, (short)0, acc01, false, false);
        acc10 = __builtin_amdgcn_wmma_f32_16x16x32_f16(
                    false, a1.h, false, b0.h, (short)0, acc10, false, false);
        acc11 = __builtin_amdgcn_wmma_f32_16x16x32_f16(
                    false, a1.h, false, b1.h, (short)0, acc11, false, false);
        __syncthreads();
    }

    // epilogue: scale by gate weight, scatter-add into q/kv buffers
    const int cl = lane & 15;
    const int mh = (lane >> 4) * 8;
#pragma unroll
    for (int r = 0; r < 8; ++r) {
        {   // M-subtile 0: rows 0..15
            const int M = mh + r;
            const float w  = wt_s[M];
            const int  row = tok_s[M];
            atomicAdd(&dst[(size_t)row * dstW + dcol0 + n0 + cl], acc00[r] * w);
            atomicAdd(&dst[(size_t)row * dstW + dcol0 + n1 + cl], acc01[r] * w);
        }
        {   // M-subtile 1: rows 16..31
            const int M = 16 + mh + r;
            const float w  = wt_s[M];
            const int  row = tok_s[M];
            atomicAdd(&dst[(size_t)row * dstW + dcol0 + n0 + cl], acc10[r] * w);
            atomicAdd(&dst[(size_t)row * dstW + dcol0 + n1 + cl], acc11[r] * w);
        }
    }
}

// ------------------------------------------------- per-token head attention
__global__ __launch_bounds__(256)
void attn_kernel(const float* __restrict__ qbuf, const float* __restrict__ kvbuf,
                 _Float16* __restrict__ att) {
    const int b = blockIdx.x;
    const int t = threadIdx.x;
    __shared__ float qs[2048], ks[2048], vs[2048];
    __shared__ float at[256];
    for (int i = t; i < 2048; i += 256) {
        qs[i] = qbuf[(size_t)b * 2048 + i];
        ks[i] = kvbuf[(size_t)b * 4096 + i];
        vs[i] = kvbuf[(size_t)b * 4096 + 2048 + i];
    }
    __syncthreads();
    {   // scores: 16x16, one per thread
        const int h = t >> 4, g = t & 15;
        float s = 0.0f;
        for (int d = 0; d < HDIM; ++d) s += qs[h * HDIM + d] * ks[g * HDIM + d];
        at[h * 16 + g] = s * ATT_SCALE;
    }
    __syncthreads();
    if (t < 16) {   // softmax over row t
        float mx = at[t * 16];
#pragma unroll
        for (int g = 1; g < 16; ++g) mx = fmaxf(mx, at[t * 16 + g]);
        float sum = 0.0f;
#pragma unroll
        for (int g = 0; g < 16; ++g) { float ev = __expf(at[t * 16 + g] - mx); at[t * 16 + g] = ev; sum += ev; }
        const float inv = 1.0f / sum;
#pragma unroll
        for (int g = 0; g < 16; ++g) at[t * 16 + g] *= inv;
    }
    __syncthreads();
    {   // out[h][d] = sum_g attn[h][g] * v[g][d]; store col = d*16 + h (swapaxes)
        const int h = t >> 4, dg = t & 15;
#pragma unroll
        for (int j = 0; j < 8; ++j) {
            const int d = dg * 8 + j;
            float s = 0.0f;
#pragma unroll
            for (int g = 0; g < 16; ++g) s += at[h * 16 + g] * vs[g * HDIM + d];
            att[(size_t)b * 2048 + d * 16 + h] = (_Float16)s;
        }
    }
}

// ------------------------------------------------ out = att @ Wp + bp (WMMA)
__global__ __launch_bounds__(256)
void proj_gemm_kernel(const _Float16* __restrict__ att, const float* __restrict__ Wp,
                      const float* __restrict__ bp, float* __restrict__ out) {
    const int colbase = blockIdx.x * 256;      // 8 tiles over 2048
    const int rowbase = blockIdx.y * 32;       // 128 tiles over 4096

    __shared__ __align__(16) _Float16 Ah[32 * 40];
    __shared__ __align__(16) _Float16 Bh[256 * 40];

    const int t = threadIdx.x;
    const int lane = t & 31;
    const int wv   = t >> 5;
    const int n0   = wv * 16;
    const int n1   = 128 + wv * 16;
    const int arow = lane & 15;
    const int akb  = (lane >> 4) * 8;
    const int bc0  = n0 + (lane & 15);
    const int bc1  = n1 + (lane & 15);
    const int bk   = (lane >> 4) * 8;

    v8f acc00 = {}; v8f acc01 = {};
    v8f acc10 = {}; v8f acc11 = {};

    for (int k0 = 0; k0 < DIMK; k0 += 32) {
        if (k0 + 32 < DIMK) {
            const int pr = t >> 3;
            const int pc = (t & 7) * 32;
            __builtin_prefetch(Wp + (size_t)(k0 + 32 + pr) * DIMK + colbase + pc, 0, 1);
        }
#pragma unroll
        for (int j = 0; j < 4; ++j) {
            const int i = t + j * 256;
            const int r = i >> 5, kk = i & 31;
            Ah[r * 40 + kk] = att[(size_t)(rowbase + r) * DIMK + k0 + kk];
        }
#pragma unroll
        for (int i = 0; i < 8; ++i) {
            const int q  = i * 256 + t;
            const int kk = q >> 6;
            const int c4 = (q & 63) << 2;
            const f32x4 v = *(const f32x4*)(Wp + (size_t)(k0 + kk) * DIMK + colbase + c4);
            Bh[(c4 + 0) * 40 + kk] = (_Float16)v.x;
            Bh[(c4 + 1) * 40 + kk] = (_Float16)v.y;
            Bh[(c4 + 2) * 40 + kk] = (_Float16)v.z;
            Bh[(c4 + 3) * 40 + kk] = (_Float16)v.w;
        }
        __syncthreads();

        Frag16 a0, a1, b0, b1;
        a0.u[0] = *(const u32x4*)(&Ah[arow * 40 + akb]);
        a0.u[1] = *(const u32x4*)(&Ah[arow * 40 + akb + 16]);
        a1.u[0] = *(const u32x4*)(&Ah[(16 + arow) * 40 + akb]);
        a1.u[1] = *(const u32x4*)(&Ah[(16 + arow) * 40 + akb + 16]);
        b0.u[0] = *(const u32x4*)(&Bh[bc0 * 40 + bk]);
        b0.u[1] = *(const u32x4*)(&Bh[bc0 * 40 + bk + 16]);
        b1.u[0] = *(const u32x4*)(&Bh[bc1 * 40 + bk]);
        b1.u[1] = *(const u32x4*)(&Bh[bc1 * 40 + bk + 16]);

        acc00 = __builtin_amdgcn_wmma_f32_16x16x32_f16(
                    false, a0.h, false, b0.h, (short)0, acc00, false, false);
        acc01 = __builtin_amdgcn_wmma_f32_16x16x32_f16(
                    false, a0.h, false, b1.h, (short)0, acc01, false, false);
        acc10 = __builtin_amdgcn_wmma_f32_16x16x32_f16(
                    false, a1.h, false, b0.h, (short)0, acc10, false, false);
        acc11 = __builtin_amdgcn_wmma_f32_16x16x32_f16(
                    false, a1.h, false, b1.h, (short)0, acc11, false, false);
        __syncthreads();
    }

    const int cl = lane & 15;
    const int mh = (lane >> 4) * 8;
#pragma unroll
    for (int r = 0; r < 8; ++r) {
        const int c0 = colbase + n0 + cl;
        const int c1 = colbase + n1 + cl;
        {
            const int M = mh + r;
            out[(size_t)(rowbase + M) * DIMK + c0] = acc00[r] + bp[c0];
            out[(size_t)(rowbase + M) * DIMK + c1] = acc01[r] + bp[c1];
        }
        {
            const int M = 16 + mh + r;
            out[(size_t)(rowbase + M) * DIMK + c0] = acc10[r] + bp[c0];
            out[(size_t)(rowbase + M) * DIMK + c1] = acc11[r] + bp[c1];
        }
    }
}

// ---------------------------------------------------------------------------
extern "C" void kernel_launch(void* const* d_in, const int* in_sizes, int n_in,
                              void* d_out, int out_size, void* d_ws, size_t ws_size,
                              hipStream_t stream) {
    const float* x   = (const float*)d_in[0];
    const float* y   = (const float*)d_in[1];
    const float* W   = (const float*)d_in[2];   // (E, DIM, 3*DIM)
    const float* Wg  = (const float*)d_in[3];   // (DIM, E)
    const float* bg  = (const float*)d_in[4];   // (E,)
    const float* Wp  = (const float*)d_in[5];   // (DIM, DIM)
    const float* bp  = (const float*)d_in[6];   // (DIM,)
    float* out       = (float*)d_out;

    // Workspace layout (bytes). Total ~118 MB.
    char* ws = (char*)d_ws;
    const size_t off_q      = 0;                                     // B*DIM f32
    const size_t off_kv     = off_q  + (size_t)BSZ * DIMK * 4;       // B*2*DIM f32
    const size_t off_counts = off_kv + (size_t)BSZ * 2 * DIMK * 4;   // 8 int
    const size_t off_cursor = off_counts + 32;                       // 8 int
    const size_t off_padoff = off_cursor + 32;                       // 8 int
    const size_t off_segex  = off_padoff + 32;                       // MAX_SEG int
    const size_t off_tok    = off_segex  + ((MAX_SEG * 4 + 31) & ~(size_t)31);
    const size_t off_wt     = off_tok + (size_t)MAX_SLOTS * 4;
    const size_t off_idx    = off_wt  + (size_t)MAX_SLOTS * 4;
    const size_t off_wgt    = off_idx + (size_t)BSZ * 2 * 4;
    const size_t off_att    = (off_wgt + (size_t)BSZ * 2 * 4 + 255) & ~(size_t)255;

    float*     qbuf      = (float*)(ws + off_q);
    float*     kvbuf     = (float*)(ws + off_kv);
    int*       counts    = (int*)  (ws + off_counts);
    int*       cursor    = (int*)  (ws + off_cursor);
    int*       pad_off   = (int*)  (ws + off_padoff);
    int*       seg_ex    = (int*)  (ws + off_segex);
    int*       tok       = (int*)  (ws + off_tok);
    float*     wt        = (float*)(ws + off_wt);
    int*       idx       = (int*)  (ws + off_idx);
    float*     wgt       = (float*)(ws + off_wgt);
    _Float16*  att       = (_Float16*)(ws + off_att);

    // 1) zero q/kv accumulation buffers + routing region (everything < off_idx)
    const long long nzero = (long long)(off_idx / 4);
    zero_kernel<<<4096, 256, 0, stream>>>((float*)ws, nzero);

    // 2) gating: logits, softmax, top-2, per-expert counts
    gate_kernel<<<BSZ, 256, 0, stream>>>(x, Wg, bg, idx, wgt, counts);

    // 3) padded per-expert offsets + segment->expert table (M=32 segments)
    offsets_kernel<<<1, 32, 0, stream>>>(counts, cursor, pad_off, seg_ex);

    // 4) scatter token assignments into per-expert slot lists
    scatter_kernel<<<BSZ / 256, 256, 0, stream>>>(idx, wgt, pad_off, cursor, tok, wt);

    // 5) grouped expert GEMM (WMMA f16): y-pass -> q cols, x-pass -> k,v cols
    moe_gemm_kernel<<<dim3(24, MAX_SEG), 256, 0, stream>>>(
        x, y, W, tok, wt, seg_ex, qbuf, kvbuf);

    // 6) per-token 16x16 head attention, writes f16 in (d*16+h) layout
    attn_kernel<<<BSZ, 256, 0, stream>>>(qbuf, kvbuf, att);

    // 7) output projection (WMMA f16) with fused bias
    proj_gemm_kernel<<<dim3(DIMK / 256, BSZ / 32), 256, 0, stream>>>(att, Wp, bp, out);
}
